// MultiHeadSelfAttention_29515015258634
// MI455X (gfx1250) — matrix operand order repaired and checked by reading, not compiled, not run on hardware
//
#include <hip/hip_runtime.h>

// ---------------------------------------------------------------------------
// Types / helpers
// ---------------------------------------------------------------------------
typedef __bf16 bf16_t;
typedef __attribute__((ext_vector_type(16))) __bf16 v16bf;
typedef __attribute__((ext_vector_type(8)))  float  v8f;

union BfBits { bf16_t b; unsigned short u; };

static __device__ __forceinline__ unsigned short f2bfbits(float f) {
  union { float f; unsigned int u; } a; a.f = f;
  unsigned int r = a.u + 0x7FFFu + ((a.u >> 16) & 1u);   // round-nearest-even
  return (unsigned short)(r >> 16);
}
static __device__ __forceinline__ bf16_t f2bf(float f) {
  BfBits o; o.u = f2bfbits(f); return o.b;
}
static __device__ __forceinline__ float bf2f(bf16_t b) {
  BfBits i; i.b = b;
  union { unsigned int u; float f; } o; o.u = ((unsigned int)i.u) << 16;
  return o.f;
}
static __device__ __forceinline__ unsigned int pack2bf(float a, float b) {
  return (unsigned int)f2bfbits(a) | ((unsigned int)f2bfbits(b) << 16);
}

union Frag {
  v16bf v;
  unsigned int u[8];
};

// two 16-byte contiguous chunks -> one 16-element bf16 fragment
static __device__ __forceinline__ void load16(Frag& f, const bf16_t* p0, const bf16_t* p1) {
  uint4 a = *(const uint4*)p0;
  uint4 b = *(const uint4*)p1;
  f.u[0] = a.x; f.u[1] = a.y; f.u[2] = a.z; f.u[3] = a.w;
  f.u[4] = b.x; f.u[5] = b.y; f.u[6] = b.z; f.u[7] = b.w;
}

static __device__ __forceinline__ v8f wmma_bf16(v16bf a, v16bf b, v8f c) {
  return __builtin_amdgcn_wmma_f32_16x16x32_bf16(false, a, false, b, (short)0, c, false, false);
}

// ---------------------------------------------------------------------------
// Problem constants
// ---------------------------------------------------------------------------
#define BATCH 8
#define SEQ   1024
#define DMODEL 1024
#define NHEAD 16
#define HDIM  64           // head dim of q/k/v
#define RDIM  128          // rotary-doubled dim of q2/k2
#define DX 21
#define DY 21
#define MX 32
#define MY 32
#define MZ 8

// ---------------------------------------------------------------------------
// fp32 -> bf16 downconvert
// ---------------------------------------------------------------------------
__global__ void __launch_bounds__(256) k_f32_to_bf16(const float* __restrict__ in,
                                                     bf16_t* __restrict__ out, int n) {
  int i = blockIdx.x * blockDim.x + threadIdx.x;
  if (i < n) out[i] = f2bf(in[i]);
}

// ---------------------------------------------------------------------------
// GEMM: C[M,N] = A[M,K] * Bw[N,K]^T   (bf16 inputs, fp32 accumulate)
// block = 256 threads = 8 waves; block tile 64x256; wave tile 32x64
// 2-deep software pipeline over K (step 64, two register fragment buffers).
// Requires K % 64 == 0.
// ---------------------------------------------------------------------------
template <bool OUT_BF16>
__global__ void __launch_bounds__(256, 1)
k_gemm_wmma(const bf16_t* __restrict__ A,
            const bf16_t* __restrict__ Bw,
            void* __restrict__ Cout,
            int M, int N, int K) {
  const int lane = threadIdx.x & 31;
  const int wid  = threadIdx.x >> 5;          // 0..7
  const int l16  = lane & 15;
  const int half = lane >> 4;

  const int mBase = blockIdx.x * 64  + (wid & 1) * 32;
  const int nBase = blockIdx.y * 256 + (wid >> 1) * 64;

  v8f acc[2][4] = {};
  Frag a[2][2], b[2][4];

  auto loadTile = [&](int buf, int k0) {
#pragma unroll
    for (int mt = 0; mt < 2; ++mt) {
      const bf16_t* ap = A + (size_t)(mBase + 16 * mt + l16) * K + k0 + half * 8;
      load16(a[buf][mt], ap, ap + 16);
    }
#pragma unroll
    for (int nt = 0; nt < 4; ++nt) {
      const bf16_t* bp = Bw + (size_t)(nBase + 16 * nt + l16) * K + k0 + half * 8;
      load16(b[buf][nt], bp, bp + 16);
    }
  };

  auto domma = [&](int buf) {
#pragma unroll
    for (int mt = 0; mt < 2; ++mt)
#pragma unroll
      for (int nt = 0; nt < 4; ++nt)
        acc[mt][nt] = wmma_bf16(a[buf][mt].v, b[buf][nt].v, acc[mt][nt]);
  };

  loadTile(0, 0);
  for (int k0 = 0; k0 < K; k0 += 64) {
    loadTile(1, k0 + 32);               // prefetch second half of this super-step
    domma(0);
    if (k0 + 64 < K) loadTile(0, k0 + 64);  // prefetch next super-step
    domma(1);
  }

#pragma unroll
  for (int mt = 0; mt < 2; ++mt)
#pragma unroll
    for (int nt = 0; nt < 4; ++nt) {
      const int col = nBase + 16 * nt + l16;
#pragma unroll
      for (int v = 0; v < 8; ++v) {
        const int row = mBase + 16 * mt + 8 * half + v;
        if (OUT_BF16)
          ((bf16_t*)Cout)[(size_t)row * N + col] = f2bf(acc[mt][nt][v]);
        else
          ((float*)Cout)[(size_t)row * N + col] = acc[mt][nt][v];
      }
    }
}

// ---------------------------------------------------------------------------
// 3D PoPE rotary: qkv_bf [B,S,3,H,64] -> q2/k2 [B,H,S,128], v [B,H,S,64]
// one thread per (b,h,s,hd)
// ---------------------------------------------------------------------------
__global__ void __launch_bounds__(256) k_pope(const bf16_t* __restrict__ qkv,
                                              const int* __restrict__ pos,
                                              const float* __restrict__ pbias,
                                              bf16_t* __restrict__ q2,
                                              bf16_t* __restrict__ k2,
                                              bf16_t* __restrict__ vout) {
  const int t  = blockIdx.x * blockDim.x + threadIdx.x;
  const int hd = t & 63;
  const int s  = (t >> 6) & 1023;
  const int h  = (t >> 16) & 15;
  const int b  = t >> 20;

  const size_t qi = ((size_t)(b * SEQ + s)) * (3 * DMODEL) + h * HDIM + hd;
  const float xq = bf2f(qkv[qi]);
  const float xk = bf2f(qkv[qi + DMODEL]);
  const float xv = bf2f(qkv[qi + 2 * DMODEL]);

  int p, dim, j, pmax;
  const int* pr = pos + (size_t)(b * SEQ + s) * 3;
  if (hd < DX)            { p = pr[0]; dim = DX;             j = hd;            pmax = MX - 1; }
  else if (hd < DX + DY)  { p = pr[1]; dim = DY;             j = hd - DX;       pmax = MY - 1; }
  else                    { p = pr[2]; dim = HDIM - DX - DY; j = hd - DX - DY;  pmax = MZ - 1; }
  p = p < 0 ? 0 : (p > pmax ? pmax : p);

  // inv_freq = BASE^(-j/dim) = exp(-(j/dim)*ln(10000))
  const float invf = __expf(-((float)j / (float)dim) * 9.210340371976184f);
  const float tt   = (float)p * invf;

  float bias = pbias[hd];
  bias = fminf(fmaxf(bias, -6.2831853071795864f), 0.0f);

  const float ct = __cosf(tt), st = __sinf(tt);
  const float cb = __cosf(bias), sb = __sinf(bias);
  const float ck = ct * cb - st * sb;
  const float sk = st * cb + ct * sb;

  const float muq = (xq > 20.0f) ? xq : log1pf(__expf(xq));
  const float muk = (xk > 20.0f) ? xk : log1pf(__expf(xk));

  const size_t row = (size_t)((b * NHEAD + h) * SEQ + s);
  q2[row * RDIM + hd]        = f2bf(muq * ct);
  q2[row * RDIM + HDIM + hd] = f2bf(muq * st);
  k2[row * RDIM + hd]        = f2bf(muk * ck);
  k2[row * RDIM + HDIM + hd] = f2bf(muk * sk);
  vout[row * HDIM + hd]      = f2bf(xv);
}

// ---------------------------------------------------------------------------
// Flash attention with transposed scores.
// grid = (S/64, B*H), block = 128 threads (4 waves, each owns 16 queries).
// Per 32-key step: S^T = K2(32x128) @ Q^T  (8 WMMAs, 2 C tiles),
// online softmax per query column (per-lane scalars), O^T += V^T @ P^T (4 WMMAs).
// V^T double-buffered in LDS (one barrier per step). K2 fragments ping-pong
// in pairs per kf sub-step (max 4 live) so loads for kf+1 overlap WMMAs of kf.
// ---------------------------------------------------------------------------
#define VT_STRIDE 36

__global__ void __launch_bounds__(128, 1)
k_attention(const bf16_t* __restrict__ q2,
            const bf16_t* __restrict__ k2,
            const bf16_t* __restrict__ vv,
            const unsigned char* __restrict__ kmask,
            bf16_t* __restrict__ attn_out) {
  __shared__ __align__(16) bf16_t VT[2][HDIM][VT_STRIDE];   // [buf][head_dim][key]

  const int lane = threadIdx.x & 31;
  const int wid  = threadIdx.x >> 5;          // 0..3
  const int l16  = lane & 15;
  const int half = lane >> 4;

  const int bh = blockIdx.y;                  // b*16 + h
  const int b  = bh >> 4;
  const int h  = bh & 15;
  const int qBase = blockIdx.x * 64 + wid * 16;

  const size_t baseQK = (size_t)bh * SEQ * RDIM;
  const size_t baseV  = (size_t)bh * SEQ * HDIM;

  // Q^T B-fragments, loaded once: lane=query col, K=feature
  Frag qf[4];
#pragma unroll
  for (int kf = 0; kf < 4; ++kf) {
    const bf16_t* qp = q2 + baseQK + (size_t)(qBase + l16) * RDIM + kf * 32 + half * 8;
    load16(qf[kf], qp, qp + 16);
  }

  const float scale = 0.08838834764831845f;   // 1/sqrt(128)
  float m_run = -3.0e38f;
  float l_run = 0.0f;
  v8f oacc[4] = {};

  // cooperative stage of V^T tile [64 hd][32 keys] into LDS buffer
  auto stageV = [&](int bufc, int kvs) {
    for (int idx = threadIdx.x; idx < 32 * 8; idx += 128) {
      const int key = idx >> 3;
      const int h8  = (idx & 7) * 8;
      union { uint4 q; bf16_t e[8]; } u;
      u.q = *(const uint4*)(vv + baseV + (size_t)(kvs + key) * HDIM + h8);
#pragma unroll
      for (int j = 0; j < 8; ++j) VT[bufc][h8 + j][key] = u.e[j];
    }
  };

  // load the two K2 A-fragments (key sub-tiles) for feature block kf
  auto loadKf = [&](Frag* dst, int kv, int kf) {
#pragma unroll
    for (int mt = 0; mt < 2; ++mt) {
      const bf16_t* kp = k2 + baseQK + (size_t)(kv + 16 * mt + l16) * RDIM + kf * 32 + half * 8;
      load16(dst[mt], kp, kp + 16);
    }
  };

  // one 32-key flash step using VT[bufc]; optionally stages VT[bufc^1] for kv+32
  auto step = [&](int bufc, int kv, bool stageNext) {
    Frag kfr[2][2];                     // ping-pong pair of K2 fragments
    loadKf(kfr[0], kv, 0);

    // overlap: stage next V tile while first K2 loads are in flight
    if (stageNext) stageV(bufc ^ 1, kv + 32);

    // ---- S^T = K2_tile(32x128) @ Q^T : pipelined over kf ----
    v8f stt[2] = {};
#pragma unroll
    for (int kf = 0; kf < 4; ++kf) {
      if (kf < 3) loadKf(kfr[(kf + 1) & 1], kv, kf + 1);
      stt[0] = wmma_bf16(kfr[kf & 1][0].v, qf[kf].v, stt[0]);
      stt[1] = wmma_bf16(kfr[kf & 1][1].v, qf[kf].v, stt[1]);
    }

    // ---- online softmax (per-lane query column), in place on stt ----
    float mtile = -3.0e38f;
#pragma unroll
    for (int mt = 0; mt < 2; ++mt)
#pragma unroll
      for (int v = 0; v < 8; ++v) {
        const int key = kv + 16 * mt + 8 * half + v;
        float s = stt[mt][v] * scale;
        if (!kmask[b * SEQ + key]) s = -3.0e38f;
        stt[mt][v] = s;
        mtile = fmaxf(mtile, s);
      }
    mtile = fmaxf(mtile, __shfl_xor(mtile, 16));
    const float m_new = fmaxf(m_run, mtile);
    const float alpha = __expf(m_run - m_new);

    float psum = 0.0f;
    unsigned int pk[2][4];
#pragma unroll
    for (int mt = 0; mt < 2; ++mt)
#pragma unroll
      for (int pp = 0; pp < 4; ++pp) {
        const float e0 = __expf(stt[mt][2 * pp]     - m_new);
        const float e1 = __expf(stt[mt][2 * pp + 1] - m_new);
        psum += e0 + e1;
        pk[mt][pp] = pack2bf(e0, e1);
      }
    psum += __shfl_xor(psum, 16);
    l_run = l_run * alpha + psum;
    m_run = m_new;
#pragma unroll
    for (int t = 0; t < 4; ++t)
#pragma unroll
      for (int v = 0; v < 8; ++v) oacc[t][v] *= alpha;

    // ---- P^T B-fragment: cross-half exchange of packed bf16 pairs ----
    Frag pf;
#pragma unroll
    for (int pp = 0; pp < 4; ++pp) {
      const unsigned int o0 = __shfl_xor(pk[0][pp], 16);
      const unsigned int o1 = __shfl_xor(pk[1][pp], 16);
      pf.u[pp]     = half ? o1        : pk[0][pp];  // keys 16*half + 2p,2p+1
      pf.u[4 + pp] = half ? pk[1][pp] : o0;         // keys 16*half+8 + 2p,2p+1
    }

    // ---- O^T += V^T(64x32) @ P^T(32x16) from LDS buffer ----
#pragma unroll
    for (int t = 0; t < 4; ++t) {
      Frag af;
      const bf16_t* vr = &VT[bufc][16 * t + l16][0];
#pragma unroll
      for (int pp = 0; pp < 8; ++pp) {
        const int kk = 16 * half + 2 * pp;
        af.u[pp] = *(const unsigned int*)(vr + kk);
      }
      oacc[t] = wmma_bf16(af.v, pf.v, oacc[t]);
    }

    __syncthreads();   // VT[bufc] reads done; VT[bufc^1] writes visible
  };

  // prologue: stage first V tile
  stageV(0, 0);
  __syncthreads();

  for (int kv = 0; kv < SEQ; kv += 64) {
    step(0, kv, true);
    step(1, kv + 32, kv + 64 < SEQ);
  }

  // ---- epilogue: normalize, write attn [B,S,D] in bf16 ----
  const float inv = 1.0f / l_run;
  const size_t orow = ((size_t)(b * SEQ + qBase + l16)) * DMODEL + h * HDIM;
#pragma unroll
  for (int t = 0; t < 4; ++t) {
    union { uint4 q; bf16_t e[8]; } u;
#pragma unroll
    for (int v = 0; v < 8; ++v) u.e[v] = f2bf(oacc[t][v] * inv);
    *(uint4*)(attn_out + orow + 16 * t + 8 * half) = u.q;
  }
}

// ---------------------------------------------------------------------------
// Launch
// ---------------------------------------------------------------------------
extern "C" void kernel_launch(void* const* d_in, const int* in_sizes, int n_in,
                              void* d_out, int out_size, void* d_ws, size_t ws_size,
                              hipStream_t stream) {
  (void)in_sizes; (void)n_in; (void)out_size;

  const float*         hs    = (const float*)d_in[0];
  const unsigned char* amask = (const unsigned char*)d_in[1];
  const int*           pos   = (const int*)d_in[2];
  const float*         wqkv  = (const float*)d_in[3];
  const float*         wout  = (const float*)d_in[4];
  const float*         pbias = (const float*)d_in[5];
  float*               out   = (float*)d_out;

  const size_t nHS   = (size_t)BATCH * SEQ * DMODEL;          // 8.39M
  const size_t nWQKV = (size_t)3 * DMODEL * DMODEL;           // 3.15M
  const size_t nWOUT = (size_t)DMODEL * DMODEL;               // 1.05M
  const size_t nQKV  = (size_t)BATCH * SEQ * 3 * DMODEL;      // 25.2M
  const size_t nR    = (size_t)BATCH * NHEAD * SEQ * RDIM;    // 16.8M
  const size_t nV    = (size_t)BATCH * NHEAD * SEQ * HDIM;    // 8.39M

  char* ws = (char*)d_ws;
  size_t off = 0;
  bf16_t* hbf    = (bf16_t*)(ws + off); off += nHS   * 2;
  bf16_t* wqkvbf = (bf16_t*)(ws + off); off += nWQKV * 2;
  bf16_t* woutbf = (bf16_t*)(ws + off); off += nWOUT * 2;
  bf16_t* qkvbf  = (bf16_t*)(ws + off); off += nQKV  * 2;
  bf16_t* q2     = (bf16_t*)(ws + off); off += nR    * 2;
  bf16_t* k2     = (bf16_t*)(ws + off); off += nR    * 2;
  bf16_t* vbf    = (bf16_t*)(ws + off); off += nV    * 2;
  bf16_t* attnbf = (bf16_t*)(ws + off); off += nHS   * 2;
  if (off > ws_size) return;  // workspace too small: bail deterministically

  // 1) downconvert inputs to bf16
  k_f32_to_bf16<<<(unsigned)((nHS   + 255) / 256), 256, 0, stream>>>(hs,   hbf,    (int)nHS);
  k_f32_to_bf16<<<(unsigned)((nWQKV + 255) / 256), 256, 0, stream>>>(wqkv, wqkvbf, (int)nWQKV);
  k_f32_to_bf16<<<(unsigned)((nWOUT + 255) / 256), 256, 0, stream>>>(wout, woutbf, (int)nWOUT);

  // 2) QKV projection: [8192,1024] x [1024,3072] -> bf16
  {
    dim3 grid(BATCH * SEQ / 64, 3 * DMODEL / 256);
    k_gemm_wmma<true><<<grid, 256, 0, stream>>>(hbf, wqkvbf, qkvbf,
                                                BATCH * SEQ, 3 * DMODEL, DMODEL);
  }

  // 3) PoPE rotary
  k_pope<<<(unsigned)(nV / 256), 256, 0, stream>>>(qkvbf, pos, pbias, q2, k2, vbf);

  // 4) attention -> attn [B,S,D] bf16
  {
    dim3 grid(SEQ / 64, BATCH * NHEAD);
    k_attention<<<grid, 128, 0, stream>>>(q2, k2, vbf, amask, attnbf);
  }

  // 5) output projection: [8192,1024] x [1024,1024] -> fp32 d_out
  {
    dim3 grid(BATCH * SEQ / 64, DMODEL / 256);
    k_gemm_wmma<false><<<grid, 256, 0, stream>>>(attnbf, woutbf, out,
                                                 BATCH * SEQ, DMODEL, DMODEL);
  }
}
